// MHAtt_36086315221735
// MI455X (gfx1250) — matrix operand order repaired and checked by reading, not compiled, not run on hardware
//
#include <hip/hip_runtime.h>

#define N_TOK 2048
#define DM    1024
#define NH    16
#define DH    64

typedef __attribute__((ext_vector_type(16))) __bf16 v16bf;
typedef __attribute__((ext_vector_type(8)))  __bf16 v8bf;
typedef __attribute__((ext_vector_type(8)))  float  v8f;

__device__ __forceinline__ v8f wmma_bf16(v16bf a, v16bf b, v8f c) {
    // D = A(16x32 bf16) * B(32x16 bf16) + C(16x16 f32)
    return __builtin_amdgcn_wmma_f32_16x16x32_bf16(
        false, a, false, b, (short)0, c, false, false);
}

__device__ __forceinline__ v16bf cat8(v8bf lo, v8bf hi) {
    v16bf r;
#pragma unroll
    for (int i = 0; i < 8; ++i) { r[i] = lo[i]; r[8 + i] = hi[i]; }
    return r;
}

// ---------------- f32 -> bf16 convert ----------------
__global__ void k_cvt(const float* __restrict__ in, __bf16* __restrict__ out, int n) {
    int i = blockIdx.x * blockDim.x + threadIdx.x;
    int stride = gridDim.x * blockDim.x;
    for (; i < n; i += stride) out[i] = (__bf16)in[i];
}

// ---- weights: convert + transpose [NH, DM, DH] -> [NH, DH, DM] (output coalesced) ----
__global__ void k_cvtT_w(const float* __restrict__ in, __bf16* __restrict__ out) {
    size_t j = (size_t)blockIdx.x * blockDim.x + threadIdx.x;  // 0..NH*DH*DM-1
    int h  = (int)(j >> 16);          // DH*DM = 65536
    int r  = (int)(j & 65535);
    int dh = r >> 10;                 // DM = 1024
    int k  = r & 1023;
    out[j] = (__bf16)in[(size_t)h * DM * DH + (size_t)k * DH + dh];
}

// ---- V_w: convert + transpose [NH, N, DH] -> [NH, DH, N] (output coalesced) ----
__global__ void k_cvtT_v(const float* __restrict__ in, __bf16* __restrict__ out) {
    size_t j = (size_t)blockIdx.x * blockDim.x + threadIdx.x;  // 0..NH*DH*N-1
    int h  = (int)(j >> 17);          // DH*N = 131072
    int r  = (int)(j & 131071);
    int dh = r >> 11;                 // N = 2048
    int n  = r & 2047;
    out[j] = (__bf16)in[(size_t)h * N_TOK * DH + (size_t)n * DH + dh];
}

// ---------------- per-head projection: X[N,DM] @ W[h][DM,DH] -> f32 [NH,N,DH] ----------------
// WbT layout: [NH, DH, DM]  => B fragment is K-contiguous per lane.
__global__ void k_proj(const __bf16* __restrict__ Xb, const __bf16* __restrict__ WbT,
                       float* __restrict__ outF) {
    const int lane = threadIdx.x & 31;
    const int wave = threadIdx.x >> 5;
    const int gw   = blockIdx.x * (blockDim.x >> 5) + wave;   // 0..2047
    const int h    = gw >> 7;                                  // head
    const int row0 = (gw & 127) * 16;
    const int hs   = lane >> 4;
    const int m    = lane & 15;

    v8f acc[4] = {};
    const __bf16* Arow = Xb + (size_t)(row0 + m) * DM;
    const __bf16* Wh   = WbT + (size_t)h * DH * DM;

    for (int k0 = 0; k0 < DM; k0 += 32) {
        v16bf a = cat8(*(const v8bf*)(Arow + k0 + 16 * hs),
                       *(const v8bf*)(Arow + k0 + 16 * hs + 8));
#pragma unroll
        for (int t = 0; t < 4; ++t) {
            // B[k][dh_col] = WT[h, dh_col, k] : contiguous 32B aligned
            v16bf b = *(const v16bf*)(Wh + (size_t)(t * 16 + m) * DM + k0 + 16 * hs);
            acc[t] = wmma_bf16(a, b, acc[t]);
        }
    }
#pragma unroll
    for (int t = 0; t < 4; ++t)
#pragma unroll
        for (int r = 0; r < 8; ++r)
            outF[((size_t)h * N_TOK + row0 + r + 8 * hs) * DH + t * 16 + m] = acc[t][r];
}

// ---------------- row L2-normalize (optional) + convert to bf16 [NH,N,DH] ----------------
__global__ void k_rownorm(const float* __restrict__ in, __bf16* __restrict__ out, int do_norm) {
    const int lane = threadIdx.x & 31;
    const int wave = threadIdx.x >> 5;
    const long row = (long)blockIdx.x * (blockDim.x >> 5) + wave;   // 0..NH*N_TOK-1
    const float* p = in + row * DH;
    float x0 = p[lane], x1 = p[lane + 32];
    float scale = 1.f;
    if (do_norm) {
        float s = x0 * x0 + x1 * x1;
#pragma unroll
        for (int off = 16; off >= 1; off >>= 1) s += __shfl_xor(s, off, 32);
        scale = 1.f / fmaxf(sqrtf(s), 1e-12f);
    }
    out[row * DH + lane]      = (__bf16)(x0 * scale);
    out[row * DH + lane + 32] = (__bf16)(x1 * scale);
}

// ---------------- logits: S[h] = Q_w[h] @ K_n[h]^T  (written to d_out) ----------------
__global__ void k_logits(const __bf16* __restrict__ Qwb, const __bf16* __restrict__ Knb,
                         float* __restrict__ S) {
    const int lane = threadIdx.x & 31;
    const int wave = threadIdx.x >> 5;
    const int gw   = blockIdx.x * (blockDim.x >> 5) + wave;   // 0..32767
    const int h    = gw >> 11;
    const int rem  = gw & 2047;
    const int nq0  = (rem >> 4) * 16;
    const int nk0  = (rem & 15) * 128;
    const int hs   = lane >> 4;
    const int m    = lane & 15;

    const __bf16* Qrow = Qwb + ((size_t)h * N_TOK + nq0 + m) * DH;
    v16bf a0 = cat8(*(const v8bf*)(Qrow + 16 * hs),      *(const v8bf*)(Qrow + 8 + 16 * hs));
    v16bf a1 = cat8(*(const v8bf*)(Qrow + 32 + 16 * hs), *(const v8bf*)(Qrow + 40 + 16 * hs));

    v8f acc[8] = {};
#pragma unroll
    for (int t = 0; t < 8; ++t) {
        const __bf16* Kcol = Knb + ((size_t)h * N_TOK + nk0 + t * 16 + m) * DH + 16 * hs;
        v16bf b0 = *(const v16bf*)(Kcol);
        v16bf b1 = *(const v16bf*)(Kcol + 32);
        acc[t] = wmma_bf16(a0, b0, acc[t]);
        acc[t] = wmma_bf16(a1, b1, acc[t]);
    }
    float* Sp = S + ((size_t)h * N_TOK + nq0) * N_TOK + nk0;
#pragma unroll
    for (int t = 0; t < 8; ++t)
#pragma unroll
        for (int r = 0; r < 8; ++r)
            Sp[(size_t)(r + 8 * hs) * N_TOK + t * 16 + m] = acc[t][r];
}

// ---------------- online softmax row stats (max, sum-exp) ----------------
__global__ void k_stats(const float* __restrict__ S, float2* __restrict__ st) {
    const int lane = threadIdx.x & 31;
    const int wave = threadIdx.x >> 5;
    const long row = (long)blockIdx.x * (blockDim.x >> 5) + wave;   // 0..NH*N_TOK-1
    const float* p = S + row * N_TOK;
    float mx = -3.0e38f, l = 0.f;
    for (int i = lane; i < N_TOK; i += 32) {
        float x  = p[i];
        float mn = fmaxf(mx, x);
        l = l * __expf(mx - mn) + __expf(x - mn);
        mx = mn;
    }
#pragma unroll
    for (int off = 16; off >= 1; off >>= 1) {
        float mo = __shfl_xor(mx, off, 32);
        float lo = __shfl_xor(l, off, 32);
        float mn = fmaxf(mx, mo);
        l  = l * __expf(mx - mn) + lo * __expf(mo - mn);
        mx = mn;
    }
    if (lane == 0) st[row] = make_float2(mx, l);
}

// ---------------- PV: A_mid[n, h*64+k] = softmax(S[h,n,:]) @ V_w[h]  (bf16 out) ----------------
// VwbT layout: [NH, DH, N] => B fragment is K-contiguous per lane.
__global__ void k_pv(const float* __restrict__ S, const float2* __restrict__ st,
                     const __bf16* __restrict__ VwbT, __bf16* __restrict__ Amid) {
    const int lane = threadIdx.x & 31;
    const int wave = threadIdx.x >> 5;
    const int gw   = blockIdx.x * (blockDim.x >> 5) + wave;   // 0..2047
    const int h    = gw >> 7;
    const int nq0  = (gw & 127) * 16;
    const int hs   = lane >> 4;
    const int m    = lane & 15;

    // A-fragment layout: each lane holds only row M=lane&15 -> row stats are lane-uniform
    float2 s2 = st[(size_t)h * N_TOK + nq0 + m];
    const float mrow = s2.x;
    const float invl = 1.f / s2.y;
    const float* Srow = S + ((size_t)h * N_TOK + nq0 + m) * N_TOK;
    const __bf16* Vh  = VwbT + (size_t)h * DH * N_TOK;

    v8f acc[4] = {};
    for (int k0 = 0; k0 < N_TOK; k0 += 32) {
        v16bf a;
#pragma unroll
        for (int e = 0; e < 8; ++e) {
            float lo = Srow[k0 + 16 * hs + e];
            float hi = Srow[k0 + 16 * hs + 8 + e];
            a[e]     = (__bf16)(__expf(lo - mrow) * invl);
            a[8 + e] = (__bf16)(__expf(hi - mrow) * invl);
        }
#pragma unroll
        for (int t = 0; t < 4; ++t) {
            // B[k][dh_col] = VwT[h, dh_col, k] : contiguous 32B aligned
            v16bf b = *(const v16bf*)(Vh + (size_t)(t * 16 + m) * N_TOK + k0 + 16 * hs);
            acc[t] = wmma_bf16(a, b, acc[t]);
        }
    }
#pragma unroll
    for (int t = 0; t < 4; ++t)
#pragma unroll
        for (int r = 0; r < 8; ++r)
            Amid[(size_t)(nq0 + r + 8 * hs) * (NH * DH) + h * DH + t * 16 + m] = (__bf16)acc[t][r];
}

// ---------------- output linear: Y = A_mid @ W0^T + b0  (f32 to d_out) ----------------
__global__ void k_out(const __bf16* __restrict__ Amid, const __bf16* __restrict__ W0b,
                      const float* __restrict__ b0, float* __restrict__ Y) {
    const int lane = threadIdx.x & 31;
    const int wave = threadIdx.x >> 5;
    const int gw   = blockIdx.x * (blockDim.x >> 5) + wave;   // 0..2047
    const int row0 = (gw >> 4) * 16;
    const int o0   = (gw & 15) * 64;
    const int hs   = lane >> 4;
    const int m    = lane & 15;

    const __bf16* Arow = Amid + (size_t)(row0 + m) * DM;
    v8f acc[4] = {};
    for (int k0 = 0; k0 < DM; k0 += 32) {
        v16bf a = cat8(*(const v8bf*)(Arow + k0 + 16 * hs),
                       *(const v8bf*)(Arow + k0 + 16 * hs + 8));
#pragma unroll
        for (int t = 0; t < 4; ++t) {
            // W0[o, k] row-major: contiguous in K -> aligned 32B vector load
            v16bf b = *(const v16bf*)(W0b + (size_t)(o0 + t * 16 + m) * DM + k0 + 16 * hs);
            acc[t] = wmma_bf16(a, b, acc[t]);
        }
    }
#pragma unroll
    for (int t = 0; t < 4; ++t) {
        float bias = b0[o0 + t * 16 + m];
#pragma unroll
        for (int r = 0; r < 8; ++r)
            Y[(size_t)(row0 + r + 8 * hs) * DM + o0 + t * 16 + m] = acc[t][r] + bias;
    }
}

extern "C" void kernel_launch(void* const* d_in, const int* in_sizes, int n_in,
                              void* d_out, int out_size, void* d_ws, size_t ws_size,
                              hipStream_t stream) {
    (void)in_sizes; (void)n_in; (void)out_size; (void)ws_size;
    const float* Q  = (const float*)d_in[0];
    const float* K  = (const float*)d_in[1];
    const float* V  = (const float*)d_in[2];
    const float* Wq = (const float*)d_in[3];
    const float* Wk = (const float*)d_in[4];
    const float* Wv = (const float*)d_in[5];
    const float* W0 = (const float*)d_in[6];
    const float* b0 = (const float*)d_in[7];

    float* S = (float*)d_out;                             // [NH, N, N] logits output
    float* Y = S + (size_t)NH * N_TOK * N_TOK;            // [N, NH*DH] final output

    // ---- workspace carve ----
    char* w = (char*)d_ws;
    const size_t ND  = (size_t)N_TOK * DM;                // 2M elements
    const size_t HDD = (size_t)NH * DM * DH;              // 1M elements
    const size_t HND = (size_t)NH * N_TOK * DH;           // 2M elements
    __bf16* Qb    = (__bf16*)w;              w += ND  * 2;
    __bf16* Kb    = (__bf16*)w;              w += ND  * 2;
    __bf16* Vb    = (__bf16*)w;              w += ND  * 2;
    __bf16* WqbT  = (__bf16*)w;              w += HDD * 2;   // [NH, DH, DM]
    __bf16* WkbT  = (__bf16*)w;              w += HDD * 2;
    __bf16* WvbT  = (__bf16*)w;              w += HDD * 2;
    __bf16* W0b   = (__bf16*)w;              w += (size_t)DM * DM * 2;
    float*  projF = (float*)w;               w += HND * 4;
    __bf16* Qwb   = (__bf16*)w;              w += HND * 2;   // [NH, N, DH]
    __bf16* Knb   = (__bf16*)w;              w += HND * 2;   // [NH, N, DH]
    __bf16* VwbT  = (__bf16*)w;              w += HND * 2;   // [NH, DH, N]
    __bf16* Amid  = (__bf16*)w;              w += ND  * 2;
    float2* stats = (float2*)w;              w += (size_t)NH * N_TOK * sizeof(float2);

    const int TPB = 256;
    // 1) convert inputs to bf16 (weights fused with transpose)
    k_cvt<<<4096, TPB, 0, stream>>>(Q,  Qb,  (int)ND);
    k_cvt<<<4096, TPB, 0, stream>>>(K,  Kb,  (int)ND);
    k_cvt<<<4096, TPB, 0, stream>>>(V,  Vb,  (int)ND);
    k_cvtT_w<<<(int)(HDD / TPB), TPB, 0, stream>>>(Wq, WqbT);
    k_cvtT_w<<<(int)(HDD / TPB), TPB, 0, stream>>>(Wk, WkbT);
    k_cvtT_w<<<(int)(HDD / TPB), TPB, 0, stream>>>(Wv, WvbT);
    k_cvt<<<4096, TPB, 0, stream>>>(W0, W0b, DM * DM);

    // 2) projections + (normalize for K) + bf16 pack (V transposed for PV B-operand)
    k_proj<<<256, TPB, 0, stream>>>(Qb, WqbT, projF);
    k_rownorm<<<4096, TPB, 0, stream>>>(projF, Qwb, 0);
    k_proj<<<256, TPB, 0, stream>>>(Kb, WkbT, projF);
    k_rownorm<<<4096, TPB, 0, stream>>>(projF, Knb, 1);
    k_proj<<<256, TPB, 0, stream>>>(Vb, WvbT, projF);
    k_cvtT_v<<<(int)(HND / TPB), TPB, 0, stream>>>(projF, VwbT);

    // 3) attention logits -> d_out
    k_logits<<<4096, TPB, 0, stream>>>(Qwb, Knb, S);

    // 4) softmax row stats
    k_stats<<<4096, TPB, 0, stream>>>(S, stats);

    // 5) P @ V_w -> A_mid (bf16)
    k_pv<<<256, TPB, 0, stream>>>(S, stats, VwbT, Amid);

    // 6) output linear -> d_out second region
    k_out<<<256, TPB, 0, stream>>>(Amid, W0b, b0, Y);
}
